// ModifiedBertSelfAttention_42563125903520
// MI455X (gfx1250) — compile-verified, matrix operand order checked by reading, not run on hardware
//
#include <hip/hip_runtime.h>

// ---------------------------------------------------------------------------
// BERT self-attention forward on MI455X (gfx1250, wave32, WMMA).
// B=4, S=2048, D=1024, H=16, DH=64.  bf16 WMMA (16x16x32), f32 accumulate.
// v3: fix async-to-LDS builtin arg types (expects v4i* per clang diagnostic).
// ---------------------------------------------------------------------------

#define B_  4
#define S_  2048
#define D_  1024
#define H_  16
#define DH_ 64

typedef __attribute__((ext_vector_type(16))) __bf16         v16bf;
typedef __attribute__((ext_vector_type(8)))  float          v8f;
typedef __attribute__((ext_vector_type(4)))  unsigned int   v4u;
typedef __attribute__((ext_vector_type(4)))  int            v4i;
typedef __attribute__((ext_vector_type(4)))  float          v4f;
typedef __attribute__((ext_vector_type(4)))  unsigned short v4us;

// fp32 -> bf16 round-to-nearest-even
__device__ __forceinline__ unsigned short f32_to_bf16_rne(float f) {
  union { float f; unsigned int u; } c; c.f = f;
  unsigned int r = (c.u + 0x7FFFu + ((c.u >> 16) & 1u)) >> 16;
  return (unsigned short)r;
}

// 16x32 bf16 fragment (A-layout per CDNA5 ISA 7.12.2):
//   lanes 0-15 : row = lane,    K = {0..7, 16..23}
//   lanes16-31 : row = lane-16, K = {8..15, 24..31}
// => two 16-byte loads per lane. Works for global or LDS pointers.
union FragAB { v16bf v; v4u q[2]; };

__device__ __forceinline__ FragAB load_frag(const unsigned short* base,
                                            int row_stride, int lane) {
  const int r = lane & 15;
  const int c = (lane >> 4) * 8;
  const unsigned short* p = base + (size_t)r * row_stride + c;
  FragAB f;
  f.q[0] = *(const v4u*)(p);
  f.q[1] = *(const v4u*)(p + 16);
  return f;
}

__device__ __forceinline__ v8f wmma_bf16(const FragAB& a, const FragAB& b, v8f c) {
  return __builtin_amdgcn_wmma_f32_16x16x32_bf16(
      /*neg_a=*/false, a.v, /*neg_b=*/false, b.v,
      /*c_mod=*/(short)0, c, /*reuse_a=*/false, /*reuse_b=*/false);
}

// --- CDNA5 async global->LDS copy (ASYNCcnt-tracked), 16B per call ----------
__device__ __forceinline__ void async_copy_b128(const unsigned short* g,
                                                unsigned short* l) {
#if defined(__has_builtin) && __has_builtin(__builtin_amdgcn_global_load_async_to_lds_b128)
  // clang signature (from diagnostic): param0/param1 are int-vec16B pointers.
  __builtin_amdgcn_global_load_async_to_lds_b128((v4i*)g, (v4i*)l, 0, 0);
#else
  *(v4u*)l = *(const v4u*)g;   // fallback: sync global_load_b128 + ds_store_b128
#endif
}

__device__ __forceinline__ void wait_async() {
#if defined(__has_builtin) && __has_builtin(__builtin_amdgcn_s_wait_asynccnt)
  __builtin_amdgcn_s_wait_asynccnt(0);
#else
  asm volatile("s_wait_asynccnt 0" ::: "memory");
#endif
}

// ---------------------------------------------------------------------------
// fp32 -> bf16 bulk convert (n divisible by 4)
// ---------------------------------------------------------------------------
__global__ void cvt_f32_to_bf16(const float* __restrict__ in,
                                unsigned short* __restrict__ out, int n) {
  int i = (blockIdx.x * blockDim.x + threadIdx.x) * 4;
  if (i + 3 < n) {
    v4f v = *(const v4f*)(in + i);
    v4us o;
    o.x = f32_to_bf16_rne(v.x);
    o.y = f32_to_bf16_rne(v.y);
    o.z = f32_to_bf16_rne(v.z);
    o.w = f32_to_bf16_rne(v.w);
    *(v4us*)(out + i) = o;
  }
}

// ---------------------------------------------------------------------------
// QKV projection: out = hidden @ W^T + b, reshaped per head.
//   z = 0: Q -> [B,H,S,DH] bf16
//   z = 1: K -> [B,H,S,DH] bf16
//   z = 2: V -> [B,H,DH,S] bf16 (transposed for attention's B-fragment loads)
// Block = 128 threads (4 waves). Wave owns 32(M) x 64(N); K-loop step 32 with
// register double-buffering so next-step loads overlap current WMMAs.
// ---------------------------------------------------------------------------
__global__ void __launch_bounds__(128)
qkv_gemm(const unsigned short* __restrict__ hb,   // [B*S, D] bf16
         const unsigned short* __restrict__ wb,   // 3 x [D, D] bf16
         const float* __restrict__ bq,
         const float* __restrict__ bk,
         const float* __restrict__ bv,
         unsigned short* __restrict__ Qb,
         unsigned short* __restrict__ Kb,
         unsigned short* __restrict__ VTb) {
  const int lane = threadIdx.x & 31;
  const int wave = threadIdx.x >> 5;
  const int z    = blockIdx.z;
  const int m0   = blockIdx.x * 128 + wave * 32;  // token rows (never cross batch)
  const int n0   = blockIdx.y * 64;               // output features (head-aligned)

  const unsigned short* W    = wb + (size_t)z * D_ * D_;
  const float*          bias = (z == 0) ? bq : (z == 1) ? bk : bv;
  const unsigned short* aB0  = hb + (size_t)m0 * D_;
  const unsigned short* aB1  = hb + (size_t)(m0 + 16) * D_;

  v8f acc[2][4];
  #pragma unroll
  for (int i = 0; i < 2; ++i)
    #pragma unroll
    for (int j = 0; j < 4; ++j) acc[i][j] = (v8f){};

  // prologue: fragments for k = 0
  FragAB a0 = load_frag(aB0, D_, lane);
  FragAB a1 = load_frag(aB1, D_, lane);
  FragAB b[4];
  #pragma unroll
  for (int j = 0; j < 4; ++j)
    b[j] = load_frag(W + (size_t)(n0 + 16 * j) * D_, D_, lane);

  for (int k = 0; k < D_; k += 32) {
    const int kn = (k + 32 < D_) ? (k + 32) : 0;  // last iter: harmless reload
    // issue next-step loads first (hidden behind the 8 WMMAs below)
    FragAB a0n = load_frag(aB0 + kn, D_, lane);
    FragAB a1n = load_frag(aB1 + kn, D_, lane);
    FragAB bn[4];
    #pragma unroll
    for (int j = 0; j < 4; ++j)
      bn[j] = load_frag(W + (size_t)(n0 + 16 * j) * D_ + kn, D_, lane);

    #pragma unroll
    for (int j = 0; j < 4; ++j) {
      acc[0][j] = wmma_bf16(a0, b[j], acc[0][j]);
      acc[1][j] = wmma_bf16(a1, b[j], acc[1][j]);
    }
    a0 = a0n; a1 = a1n;
    #pragma unroll
    for (int j = 0; j < 4; ++j) b[j] = bn[j];
  }

  // C/D layout: lane&15 = column, VGPR r = row (upper half-wave: row r+8)
  const int colL  = lane & 15;
  const int rbase = (lane >> 4) * 8;

  #pragma unroll
  for (int i = 0; i < 2; ++i) {
    const int mb    = m0 + 16 * i + rbase;
    const int bidx  = mb / S_;
    const int sbase = mb % S_;
    #pragma unroll
    for (int j = 0; j < 4; ++j) {
      const int   n    = n0 + 16 * j + colL;
      const float bval = bias[n];
      const int   h    = n >> 6;
      const int   dh   = n & 63;
      if (z == 2) {
        // V^T: lane writes 8 consecutive S positions at fixed dh -> one b128 store
        union { unsigned short s[8]; v4u q; } pk;
        #pragma unroll
        for (int r = 0; r < 8; ++r) pk.s[r] = f32_to_bf16_rne(acc[i][j][r] + bval);
        unsigned short* dst =
            VTb + ((((size_t)bidx * H_ + h) * DH_ + dh) * S_ + sbase);
        *(v4u*)dst = pk.q;
      } else {
        unsigned short* dst = (z == 0) ? Qb : Kb;
        #pragma unroll
        for (int r = 0; r < 8; ++r) {
          dst[(((size_t)bidx * H_ + h) * S_ + (sbase + r)) * DH_ + dh] =
              f32_to_bf16_rne(acc[i][j][r] + bval);
        }
      }
    }
  }
}

// ---------------------------------------------------------------------------
// Flash attention. Block = 128 threads (4 waves) sharing one (b,h); wave w
// owns 16 query rows. Per 32-key step the block cooperatively stages the
// K tile (32x64 bf16, 4KB, contiguous) and V^T tile (64x32 bf16, 4KB) into
// double-buffered LDS via async global->LDS copies; s_wait_asynccnt + one
// barrier per step; next tile's DMA overlaps current tile's WMMAs/softmax.
// ---------------------------------------------------------------------------
__device__ __forceinline__ void stage_tiles(const unsigned short* __restrict__ Kh,
                                            const unsigned short* __restrict__ Vt,
                                            unsigned short* kb, unsigned short* vb,
                                            int kt, int t) {
  // K tile: rows kt..kt+31 x DH, fully contiguous (2048 elems). 2 chunks/thread.
  const unsigned short* kg = Kh + (size_t)kt * DH_;
  async_copy_b128(kg + t * 8,        kb + t * 8);
  async_copy_b128(kg + 1024 + t * 8, kb + 1024 + t * 8);
  // V^T tile: 64 rows (dh), 32 keys each (64B/row), global row stride S_.
  {
    const int r0 = t >> 2, c0 = (t & 3) * 8;
    async_copy_b128(Vt + (size_t)r0 * S_ + kt + c0, vb + r0 * 32 + c0);
    const int i1 = t + 128;
    const int r1 = i1 >> 2, c1 = (i1 & 3) * 8;
    async_copy_b128(Vt + (size_t)r1 * S_ + kt + c1, vb + r1 * 32 + c1);
  }
}

__global__ void __launch_bounds__(128)
flash_attn(const unsigned short* __restrict__ Qb,   // [B,H,S,DH]
           const unsigned short* __restrict__ Kb,   // [B,H,S,DH]
           const unsigned short* __restrict__ VTb,  // [B,H,DH,S]
           float* __restrict__ out) {               // [B,S,D]
  __shared__ __align__(16) unsigned short kbuf[2][32 * DH_];  // 2 x 4KB
  __shared__ __align__(16) unsigned short vbuf[2][DH_ * 32];  // 2 x 4KB
  __shared__ __align__(16) unsigned short pstage[4][16 * 32]; // 4 x 1KB

  const int t    = threadIdx.x;
  const int lane = t & 31;
  const int wave = t >> 5;
  const int bh   = blockIdx.x;                  // b*H + h
  const int q0   = blockIdx.y * 64 + wave * 16;
  const int bI   = bh / H_;
  const int h    = bh % H_;

  const unsigned short* Qh = Qb  + (size_t)bh * S_ * DH_;
  const unsigned short* Kh = Kb  + (size_t)bh * S_ * DH_;
  const unsigned short* Vt = VTb + (size_t)bh * DH_ * S_;

  const FragAB qa0 = load_frag(Qh + (size_t)q0 * DH_ + 0,  DH_, lane);
  const FragAB qa1 = load_frag(Qh + (size_t)q0 * DH_ + 32, DH_, lane);

  float mrow[8], lrow[8];
  v8f   o[4];
  #pragma unroll
  for (int r = 0; r < 8; ++r) { mrow[r] = -1e30f; lrow[r] = 0.0f; }
  #pragma unroll
  for (int j = 0; j < 4; ++j) o[j] = (v8f){};

  const float scale = 0.125f;  // 1/sqrt(DH)
  unsigned short* pw = pstage[wave];
  const int colL  = lane & 15;
  const int rbase = (lane >> 4) * 8;

  int p = 0;
  stage_tiles(Kh, Vt, kbuf[0], vbuf[0], 0, t);

  for (int kt = 0; kt < S_; kt += 32) {
    wait_async();        // drain my async copies into buffer p
    __syncthreads();     // all waves: buffer p populated, buffer p^1 free
    if (kt + 32 < S_)
      stage_tiles(Kh, Vt, kbuf[p ^ 1], vbuf[p ^ 1], kt + 32, t);

    const unsigned short* kc = kbuf[p];
    const unsigned short* vc = vbuf[p];

    const FragAB kb00 = load_frag(kc + 0,             DH_, lane);
    const FragAB kb01 = load_frag(kc + 32,            DH_, lane);
    const FragAB kb10 = load_frag(kc + 16 * DH_,      DH_, lane);
    const FragAB kb11 = load_frag(kc + 16 * DH_ + 32, DH_, lane);

    v8f s0 = (v8f){}, s1 = (v8f){};
    s0 = wmma_bf16(qa0, kb00, s0);
    s0 = wmma_bf16(qa1, kb01, s0);
    s1 = wmma_bf16(qa0, kb10, s1);
    s1 = wmma_bf16(qa1, kb11, s1);

    #pragma unroll
    for (int r = 0; r < 8; ++r) {
      const float a = s0[r] * scale;
      const float c = s1[r] * scale;
      float mx = fmaxf(a, c);
      mx = fmaxf(mx, __shfl_xor(mx, 8, 32));   // stays within 16-lane half
      mx = fmaxf(mx, __shfl_xor(mx, 4, 32));
      mx = fmaxf(mx, __shfl_xor(mx, 2, 32));
      mx = fmaxf(mx, __shfl_xor(mx, 1, 32));
      const float mnew = fmaxf(mrow[r], mx);
      const float fsc  = __expf(mrow[r] - mnew);
      mrow[r] = mnew;
      lrow[r] *= fsc;
      #pragma unroll
      for (int j = 0; j < 4; ++j) o[j][r] *= fsc;
      const float p0 = __expf(a - mnew);
      const float p1 = __expf(c - mnew);
      lrow[r] += p0 + p1;                      // lane-partial row sum
      // C-layout -> row-major 16x32 staging tile (wave-local, DScnt-ordered)
      pw[(rbase + r) * 32 + colL]      = f32_to_bf16_rne(p0);
      pw[(rbase + r) * 32 + 16 + colL] = f32_to_bf16_rne(p1);
    }

    // reload probs in A-fragment layout
    FragAB pa;
    {
      const int r = lane & 15;
      const int c = (lane >> 4) * 8;
      pa.q[0] = *(const v4u*)(pw + r * 32 + c);
      pa.q[1] = *(const v4u*)(pw + r * 32 + c + 16);
    }

    #pragma unroll
    for (int j = 0; j < 4; ++j) {
      const FragAB vbf = load_frag(vc + 16 * j * 32, 32, lane);
      o[j] = wmma_bf16(pa, vbf, o[j]);
    }
    p ^= 1;
  }

  // normalize and write ctx -> out[b, s, h*64 + dh] (fp32)
  #pragma unroll
  for (int r = 0; r < 8; ++r) {
    float l = lrow[r];
    l += __shfl_xor(l, 8, 32);
    l += __shfl_xor(l, 4, 32);
    l += __shfl_xor(l, 2, 32);
    l += __shfl_xor(l, 1, 32);
    const float inv = 1.0f / l;
    const int s = q0 + rbase + r;
    float* dst = out + (((size_t)bI * S_ + s) * D_ + h * DH_);
    #pragma unroll
    for (int j = 0; j < 4; ++j) dst[16 * j + colL] = o[j][r] * inv;
  }
}

// ---------------------------------------------------------------------------
extern "C" void kernel_launch(void* const* d_in, const int* in_sizes, int n_in,
                              void* d_out, int out_size, void* d_ws, size_t ws_size,
                              hipStream_t stream) {
  const float* hidden = (const float*)d_in[0];
  const float* Wq     = (const float*)d_in[1];
  const float* bq     = (const float*)d_in[2];
  const float* Wk     = (const float*)d_in[3];
  const float* bk     = (const float*)d_in[4];
  const float* Wv     = (const float*)d_in[5];
  const float* bv     = (const float*)d_in[6];
  float* out = (float*)d_out;

  const size_t nHid = (size_t)B_ * S_ * D_;  // 8,388,608
  const size_t nW   = (size_t)D_ * D_;       // 1,048,576

  unsigned short* ws  = (unsigned short*)d_ws;
  unsigned short* hb  = ws;           // hidden bf16
  unsigned short* wqb = hb  + nHid;   // 3 weight matrices, contiguous
  unsigned short* wkb = wqb + nW;
  unsigned short* wvb = wkb + nW;
  unsigned short* Qb  = wvb + nW;     // [B,H,S,DH]
  unsigned short* Kb  = Qb  + nHid;   // [B,H,S,DH]
  unsigned short* VTb = Kb  + nHid;   // [B,H,DH,S]

  cvt_f32_to_bf16<<<(unsigned)(nHid / 1024), 256, 0, stream>>>(hidden, hb, (int)nHid);
  cvt_f32_to_bf16<<<(unsigned)(nW   / 1024), 256, 0, stream>>>(Wq, wqb, (int)nW);
  cvt_f32_to_bf16<<<(unsigned)(nW   / 1024), 256, 0, stream>>>(Wk, wkb, (int)nW);
  cvt_f32_to_bf16<<<(unsigned)(nW   / 1024), 256, 0, stream>>>(Wv, wvb, (int)nW);

  dim3 gg(B_ * S_ / 128, D_ / 64, 3);
  qkv_gemm<<<gg, 128, 0, stream>>>(hb, wqb, bq, bk, bv, Qb, Kb, VTb);

  dim3 ga(B_ * H_, S_ / 64);
  flash_attn<<<ga, 128, 0, stream>>>(Qb, Kb, VTb, out);
}
